// LocalCausalAttention_78864189489616
// MI455X (gfx1250) — matrix operand order repaired and checked by reading, not compiled, not run on hardware
//
#include <hip/hip_runtime.h>

typedef __attribute__((ext_vector_type(16))) __bf16 v16bf;
typedef __attribute__((ext_vector_type(8)))  __bf16 v8bf;
typedef __attribute__((ext_vector_type(8)))  float  v8f;

#define BDIM 4
#define TDIM 4096
#define HDIM 1024
#define WIN  8            // lookback window; 9 taps
#define TM   56           // output rows per block
#define ROWS 64           // TM + WIN halo (4 row-tiles)
#define HC   128          // output-H chunk (8 col-tiles)
#define KC   32           // WMMA K depth (bf16)
#define NTILES 74         // ceil(4096/56)
#define SSTR 40           // bf16 LDS stride for staged k-tiles (80B rows, 16B aligned)
#define QBSTR 136         // bf16 LDS stride for Q/K/V chunks (272B rows, 16B aligned)
#define NTHREADS 512

#define STATES_N (BDIM * TDIM * HDIM)      // 16,777,216
#define W_N      (HDIM * HDIM)             // 1,048,576
#define WS_NEEDED ((size_t)(STATES_N + 3 * W_N) * 2)

// -------- async kernel LDS: staging (double-buffered) overlaid with chunk buffers --------
struct SharedAsync {
    union {
        struct {                       // GEMM phase (kc loop)
            __bf16 sS [2][ROWS * SSTR];   // 2 x  5120 B
            __bf16 sW0[2][HC   * SSTR];   // 2 x 10240 B
            __bf16 sW1[2][HC   * SSTR];   // 2 x 10240 B
        } g;                              // 51200 B
        struct {                       // score / output phase
            __bf16 sQb[ROWS * QBSTR];     // 17408 B
            __bf16 sKb[ROWS * QBSTR];     // 17408 B
        } a;                              // 34816 B
    } u;
    float sc[TM * 12];                    //  2688 B
};                                        // 53888 B total

// -------- fallback kernel LDS (round-2 layout) --------
struct SharedFB {
    __bf16 sS [ROWS * SSTR];
    __bf16 sW0[HC   * SSTR];
    __bf16 sW1[HC   * SSTR];
    __bf16 sQb[ROWS * QBSTR];
    __bf16 sKb[ROWS * QBSTR];
    float  sc [TM * 12];
};                                        // 63104 B

__device__ __forceinline__ void cvt_store4(__bf16* d, float4 s) {
    d[0] = (__bf16)s.x; d[1] = (__bf16)s.y; d[2] = (__bf16)s.z; d[3] = (__bf16)s.w;
}
// A fragment (16-bit A 16x32): K contiguous 16 per lane-half, K = {b..b+15}
__device__ __forceinline__ v16bf ld_frag_A(const __bf16* row, int kA) {
    v8bf lo = *(const v8bf*)(row + kA);
    v8bf hi = *(const v8bf*)(row + kA + 8);
    return __builtin_shufflevector(lo, hi, 0,1,2,3,4,5,6,7,8,9,10,11,12,13,14,15);
}
// B fragment (32x16): lane holds column N=lane%16; K contiguous 16 per lane-half
__device__ __forceinline__ v16bf ld_frag_B(const __bf16* row, int kB) {
    v8bf lo = *(const v8bf*)(row + kB);
    v8bf hi = *(const v8bf*)(row + kB + 8);
    return __builtin_shufflevector(lo, hi, 0,1,2,3,4,5,6,7,8,9,10,11,12,13,14,15);
}

// ---- CDNA5 async global->LDS copy (ASYNCcnt path), 16B per lane ----
__device__ __forceinline__ void async_ld16(void* lds, const void* g) {
    asm volatile("global_load_async_to_lds_b128 %0, %1, off"
                 :: "v"((unsigned)(uintptr_t)lds), "v"(g) : "memory");
}
__device__ __forceinline__ void wait_async0() {
    asm volatile("s_wait_asynccnt 0x0" ::: "memory");
}

// ---------------- prep: f32 -> bf16 cast ----------------
__global__ __launch_bounds__(256)
void cvt_f32_bf16(const float* __restrict__ src, __bf16* __restrict__ dst, int n) {
    const int i = (blockIdx.x * 256 + threadIdx.x) * 4;
    if (i < n) {
        const float4 v = *(const float4*)(src + i);
        cvt_store4(dst + i, v);
    }
}

// ---------------- main fused kernel (bf16 inputs, pipelined async staging) ----------------
__global__ __launch_bounds__(NTHREADS)
void fused_local_attn_async(const __bf16* __restrict__ sbf,
                            const __bf16* __restrict__ WqB, const float* __restrict__ bq,
                            const __bf16* __restrict__ WkB, const float* __restrict__ bk,
                            const __bf16* __restrict__ WvB, const float* __restrict__ bv,
                            float* __restrict__ out)
{
    __shared__ SharedAsync sm;

    const int tid   = threadIdx.x;
    const int lane  = tid & 31;
    const int wave  = tid >> 5;
    const int rtile = wave & 3;
    const int cpair = wave >> 2;
    const int b     = blockIdx.x / NTILES;
    const int t0    = (blockIdx.x % NTILES) * TM;

    // S staging: threads 0..255 copy 16B each (64 rows x 4 chunks)
    const int srow = (tid >> 2) & 63;
    const int sch  = (tid & 3) * 8;
    int grow = t0 + srow - WIN;
    grow = grow < 0 ? 0 : (grow > TDIM - 1 ? TDIM - 1 : grow);
    const __bf16* sptr = sbf + ((size_t)b * TDIM + grow) * HDIM + sch;
    // W staging: all 512 threads copy 16B each (128 rows x 4 chunks)
    const int wr  = tid >> 2;
    const int wch = (tid & 3) * 8;

    const int lM    = lane & 15;
    const int lhalf = lane >> 4;
    const int kA    = lhalf * 16;
    const int kB    = lhalf * 16;
    const int col0  = cpair * 32 + lM;

    for (int i = tid; i < TM * 12; i += NTHREADS) sm.sc[i] = 0.f;
    __syncthreads();

#define ISSUE_QK(bufi, kcv) do {                                                  \
        if (tid < 256) async_ld16(&sm.u.g.sS[bufi][srow * SSTR + sch], sptr + (kcv)); \
        async_ld16(&sm.u.g.sW0[bufi][wr * SSTR + wch], wq + (kcv));               \
        async_ld16(&sm.u.g.sW1[bufi][wr * SSTR + wch], wk + (kcv));               \
    } while (0)
#define ISSUE_V(bufi, kcv) do {                                                   \
        if (tid < 256) async_ld16(&sm.u.g.sS[bufi][srow * SSTR + sch], sptr + (kcv)); \
        async_ld16(&sm.u.g.sW0[bufi][wr * SSTR + wch], wv + (kcv));               \
    } while (0)

    // -------- Pass 1: Q,K projection + banded scores --------
    for (int hc = 0; hc < HDIM; hc += HC) {
        const __bf16* wq = WqB + (size_t)(hc + wr) * HDIM + wch;
        const __bf16* wk = WkB + (size_t)(hc + wr) * HDIM + wch;
        v8f accQ0 = {}, accQ1 = {}, accK0 = {}, accK1 = {};

        ISSUE_QK(0, 0);                       // pipeline prologue
        for (int kc = 0; kc < HDIM; kc += KC) {
            const int cur = (kc >> 5) & 1;
            wait_async0();
            __syncthreads();                  // buf[cur] ready; prior reads finished
            if (kc + KC < HDIM) ISSUE_QK(cur ^ 1, kc + KC);   // DMA overlaps WMMAs

            const v16bf af  = ld_frag_A(&sm.u.g.sS [cur][(rtile * 16 + lM) * SSTR], kA);
            const v16bf bq0 = ld_frag_B(&sm.u.g.sW0[cur][(cpair * 32 + lM) * SSTR], kB);
            const v16bf bq1 = ld_frag_B(&sm.u.g.sW0[cur][(cpair * 32 + 16 + lM) * SSTR], kB);
            const v16bf bk0 = ld_frag_B(&sm.u.g.sW1[cur][(cpair * 32 + lM) * SSTR], kB);
            const v16bf bk1 = ld_frag_B(&sm.u.g.sW1[cur][(cpair * 32 + 16 + lM) * SSTR], kB);

            accQ0 = __builtin_amdgcn_wmma_f32_16x16x32_bf16(false, af, false, bq0, (short)0, accQ0, false, false);
            accQ1 = __builtin_amdgcn_wmma_f32_16x16x32_bf16(false, af, false, bq1, (short)0, accQ1, false, false);
            accK0 = __builtin_amdgcn_wmma_f32_16x16x32_bf16(false, af, false, bk0, (short)0, accK0, false, false);
            accK1 = __builtin_amdgcn_wmma_f32_16x16x32_bf16(false, af, false, bk1, (short)0, accK1, false, false);
        }
        __syncthreads();                      // union overlay: GEMM phase -> chunk phase

        const float bq0s = bq[hc + col0], bq1s = bq[hc + col0 + 16];
        const float bk0s = bk[hc + col0], bk1s = bk[hc + col0 + 16];
#pragma unroll
        for (int r = 0; r < 8; ++r) {
            const int m = rtile * 16 + r + lhalf * 8;
            sm.u.a.sQb[m * QBSTR + col0]      = (__bf16)(accQ0[r] + bq0s);
            sm.u.a.sQb[m * QBSTR + col0 + 16] = (__bf16)(accQ1[r] + bq1s);
            sm.u.a.sKb[m * QBSTR + col0]      = (__bf16)(accK0[r] + bk0s);
            sm.u.a.sKb[m * QBSTR + col0 + 16] = (__bf16)(accK1[r] + bk1s);
        }
        __syncthreads();

        if (tid < TM * 9) {
            const int ot = tid / 9, w = tid % 9;
            const v8bf* qr = (const v8bf*)&sm.u.a.sQb[(ot + WIN) * QBSTR];
            const v8bf* kr = (const v8bf*)&sm.u.a.sKb[(ot + w) * QBSTR];
            float s = 0.f;
#pragma unroll 4
            for (int h8 = 0; h8 < HC / 8; ++h8) {
                const v8bf qv = qr[h8], kv = kr[h8];
#pragma unroll
                for (int j = 0; j < 8; ++j) s += (float)qv[j] * (float)kv[j];
            }
            sm.sc[ot * 12 + w] += s;
        }
        __syncthreads();                      // chunk phase done before next hc staging
    }

    // -------- softmax over 9 taps --------
    if (tid < TM) {
        const int gt = t0 + tid;
        float v[9];
        float mx = -__builtin_inff();
#pragma unroll
        for (int w = 0; w < 9; ++w) {
            const bool valid = (gt - WIN + w) >= 0;
            const float s = valid ? sm.sc[tid * 12 + w] * 0.03125f : -__builtin_inff();
            v[w] = s; mx = fmaxf(mx, s);
        }
        float sum = 0.f;
#pragma unroll
        for (int w = 0; w < 9; ++w) { const float e = __expf(v[w] - mx); v[w] = e; sum += e; }
        const float inv = 1.f / sum;
#pragma unroll
        for (int w = 0; w < 9; ++w) sm.sc[tid * 12 + w] = v[w] * inv;
    }
    __syncthreads();

    // -------- Pass 2: V projection + weighted output --------
    for (int hc = 0; hc < HDIM; hc += HC) {
        const __bf16* wv = WvB + (size_t)(hc + wr) * HDIM + wch;
        v8f accV0 = {}, accV1 = {};

        ISSUE_V(0, 0);
        for (int kc = 0; kc < HDIM; kc += KC) {
            const int cur = (kc >> 5) & 1;
            wait_async0();
            __syncthreads();
            if (kc + KC < HDIM) ISSUE_V(cur ^ 1, kc + KC);

            const v16bf af  = ld_frag_A(&sm.u.g.sS [cur][(rtile * 16 + lM) * SSTR], kA);
            const v16bf bv0 = ld_frag_B(&sm.u.g.sW0[cur][(cpair * 32 + lM) * SSTR], kB);
            const v16bf bv1 = ld_frag_B(&sm.u.g.sW0[cur][(cpair * 32 + 16 + lM) * SSTR], kB);
            accV0 = __builtin_amdgcn_wmma_f32_16x16x32_bf16(false, af, false, bv0, (short)0, accV0, false, false);
            accV1 = __builtin_amdgcn_wmma_f32_16x16x32_bf16(false, af, false, bv1, (short)0, accV1, false, false);
        }
        __syncthreads();                      // union overlay boundary

        const float bv0s = bv[hc + col0], bv1s = bv[hc + col0 + 16];
#pragma unroll
        for (int r = 0; r < 8; ++r) {
            const int m = rtile * 16 + r + lhalf * 8;
            sm.u.a.sKb[m * QBSTR + col0]      = (__bf16)(accV0[r] + bv0s);
            sm.u.a.sKb[m * QBSTR + col0 + 16] = (__bf16)(accV1[r] + bv1s);
        }
        __syncthreads();

        for (int i = tid; i < TM * HC; i += NTHREADS) {
            const int ot = i >> 7, col = i & (HC - 1);
            const float* ar = &sm.sc[ot * 12];
            float o = 0.f;
#pragma unroll
            for (int w = 0; w < 9; ++w) o += ar[w] * (float)sm.u.a.sKb[(ot + w) * QBSTR + col];
            const int gt = t0 + ot;
            if (gt < TDIM) out[((size_t)b * TDIM + gt) * HDIM + hc + col] = o;
        }
        __syncthreads();
    }
#undef ISSUE_QK
#undef ISSUE_V
}

// ---------------- fallback (f32 inputs, VGPR staging) ----------------
__global__ __launch_bounds__(NTHREADS)
void fused_local_attn_fb(const float* __restrict__ states,
                         const float* __restrict__ Wq, const float* __restrict__ bq,
                         const float* __restrict__ Wk, const float* __restrict__ bk,
                         const float* __restrict__ Wv, const float* __restrict__ bv,
                         float* __restrict__ out)
{
    __shared__ SharedFB sm;
    const int tid   = threadIdx.x;
    const int lane  = tid & 31;
    const int wave  = tid >> 5;
    const int rtile = wave & 3;
    const int cpair = wave >> 2;
    const int b     = blockIdx.x / NTILES;
    const int t0    = (blockIdx.x % NTILES) * TM;

    const int srow = tid >> 3;
    const int scol = (tid & 7) * 4;
    int grow = t0 + srow - WIN;
    grow = grow < 0 ? 0 : (grow > TDIM - 1 ? TDIM - 1 : grow);
    const float* sptr = states + ((size_t)b * TDIM + grow) * HDIM;

    const int lM    = lane & 15;
    const int lhalf = lane >> 4;
    const int kA    = lhalf * 16;
    const int kB    = lhalf * 16;
    const int col0  = cpair * 32 + lM;

    for (int i = tid; i < TM * 12; i += NTHREADS) sm.sc[i] = 0.f;
    __syncthreads();

    for (int hc = 0; hc < HDIM; hc += HC) {
        v8f accQ0 = {}, accQ1 = {}, accK0 = {}, accK1 = {};
        for (int kc = 0; kc < HDIM; kc += KC) {
            cvt_store4(&sm.sS[srow * SSTR + scol], *(const float4*)(sptr + kc + scol));
#pragma unroll
            for (int i = 0; i < 2; ++i) {
                const int j = tid + NTHREADS * i;
                const int wr2 = j >> 3, wc = (j & 7) * 4;
                cvt_store4(&sm.sW0[wr2 * SSTR + wc], *(const float4*)(Wq + (size_t)(hc + wr2) * HDIM + kc + wc));
                cvt_store4(&sm.sW1[wr2 * SSTR + wc], *(const float4*)(Wk + (size_t)(hc + wr2) * HDIM + kc + wc));
            }
            __syncthreads();
            const v16bf af  = ld_frag_A(&sm.sS [(rtile * 16 + lM) * SSTR], kA);
            const v16bf bq0 = ld_frag_B(&sm.sW0[(cpair * 32 + lM) * SSTR], kB);
            const v16bf bq1 = ld_frag_B(&sm.sW0[(cpair * 32 + 16 + lM) * SSTR], kB);
            const v16bf bk0 = ld_frag_B(&sm.sW1[(cpair * 32 + lM) * SSTR], kB);
            const v16bf bk1 = ld_frag_B(&sm.sW1[(cpair * 32 + 16 + lM) * SSTR], kB);
            accQ0 = __builtin_amdgcn_wmma_f32_16x16x32_bf16(false, af, false, bq0, (short)0, accQ0, false, false);
            accQ1 = __builtin_amdgcn_wmma_f32_16x16x32_bf16(false, af, false, bq1, (short)0, accQ1, false, false);
            accK0 = __builtin_amdgcn_wmma_f32_16x16x32_bf16(false, af, false, bk0, (short)0, accK0, false, false);
            accK1 = __builtin_amdgcn_wmma_f32_16x16x32_bf16(false, af, false, bk1, (short)0, accK1, false, false);
            __syncthreads();
        }
        const float bq0s = bq[hc + col0], bq1s = bq[hc + col0 + 16];
        const float bk0s = bk[hc + col0], bk1s = bk[hc + col0 + 16];
#pragma unroll
        for (int r = 0; r < 8; ++r) {
            const int m = rtile * 16 + r + lhalf * 8;
            sm.sQb[m * QBSTR + col0]      = (__bf16)(accQ0[r] + bq0s);
            sm.sQb[m * QBSTR + col0 + 16] = (__bf16)(accQ1[r] + bq1s);
            sm.sKb[m * QBSTR + col0]      = (__bf16)(accK0[r] + bk0s);
            sm.sKb[m * QBSTR + col0 + 16] = (__bf16)(accK1[r] + bk1s);
        }
        __syncthreads();
        if (tid < TM * 9) {
            const int ot = tid / 9, w = tid % 9;
            const v8bf* qr = (const v8bf*)&sm.sQb[(ot + WIN) * QBSTR];
            const v8bf* kr = (const v8bf*)&sm.sKb[(ot + w) * QBSTR];
            float s = 0.f;
#pragma unroll 4
            for (int h8 = 0; h8 < HC / 8; ++h8) {
                const v8bf qv = qr[h8], kv = kr[h8];
#pragma unroll
                for (int j = 0; j < 8; ++j) s += (float)qv[j] * (float)kv[j];
            }
            sm.sc[ot * 12 + w] += s;
        }
        __syncthreads();
    }

    if (tid < TM) {
        const int gt = t0 + tid;
        float v[9];
        float mx = -__builtin_inff();
#pragma unroll
        for (int w = 0; w < 9; ++w) {
            const bool valid = (gt - WIN + w) >= 0;
            const float s = valid ? sm.sc[tid * 12 + w] * 0.03125f : -__builtin_inff();
            v[w] = s; mx = fmaxf(mx, s);
        }
        float sum = 0.f;
#pragma unroll
        for (int w = 0; w < 9; ++w) { const float e = __expf(v[w] - mx); v[w] = e; sum += e; }
        const float inv = 1.f / sum;
#pragma unroll
        for (int w = 0; w < 9; ++w) sm.sc[tid * 12 + w] = v[w] * inv;
    }
    __syncthreads();

    for (int hc = 0; hc < HDIM; hc += HC) {
        v8f accV0 = {}, accV1 = {};
        for (int kc = 0; kc < HDIM; kc += KC) {
            cvt_store4(&sm.sS[srow * SSTR + scol], *(const float4*)(sptr + kc + scol));
#pragma unroll
            for (int i = 0; i < 2; ++i) {
                const int j = tid + NTHREADS * i;
                const int wr2 = j >> 3, wc = (j & 7) * 4;
                cvt_store4(&sm.sW0[wr2 * SSTR + wc], *(const float4*)(Wv + (size_t)(hc + wr2) * HDIM + kc + wc));
            }
            __syncthreads();
            const v16bf af  = ld_frag_A(&sm.sS [(rtile * 16 + lM) * SSTR], kA);
            const v16bf bv0 = ld_frag_B(&sm.sW0[(cpair * 32 + lM) * SSTR], kB);
            const v16bf bv1 = ld_frag_B(&sm.sW0[(cpair * 32 + 16 + lM) * SSTR], kB);
            accV0 = __builtin_amdgcn_wmma_f32_16x16x32_bf16(false, af, false, bv0, (short)0, accV0, false, false);
            accV1 = __builtin_amdgcn_wmma_f32_16x16x32_bf16(false, af, false, bv1, (short)0, accV1, false, false);
            __syncthreads();
        }
        const float bv0s = bv[hc + col0], bv1s = bv[hc + col0 + 16];
#pragma unroll
        for (int r = 0; r < 8; ++r) {
            const int m = rtile * 16 + r + lhalf * 8;
            sm.sKb[m * QBSTR + col0]      = (__bf16)(accV0[r] + bv0s);
            sm.sKb[m * QBSTR + col0 + 16] = (__bf16)(accV1[r] + bv1s);
        }
        __syncthreads();
        for (int i = tid; i < TM * HC; i += NTHREADS) {
            const int ot = i >> 7, col = i & (HC - 1);
            const float* ar = &sm.sc[ot * 12];
            float o = 0.f;
#pragma unroll
            for (int w = 0; w < 9; ++w) o += ar[w] * (float)sm.sKb[(ot + w) * QBSTR + col];
            const int gt = t0 + ot;
            if (gt < TDIM) out[((size_t)b * TDIM + gt) * HDIM + hc + col] = o;
        }
        __syncthreads();
    }
}

extern "C" void kernel_launch(void* const* d_in, const int* in_sizes, int n_in,
                              void* d_out, int out_size, void* d_ws, size_t ws_size,
                              hipStream_t stream) {
    const float* states = (const float*)d_in[0];
    const float* Wq     = (const float*)d_in[1];
    const float* bq     = (const float*)d_in[2];
    const float* Wk     = (const float*)d_in[3];
    const float* bk     = (const float*)d_in[4];
    const float* Wv     = (const float*)d_in[5];
    const float* bv     = (const float*)d_in[6];
    float* out = (float*)d_out;

    dim3 grid(BDIM * NTILES);
    dim3 block(NTHREADS);

    if (ws_size >= WS_NEEDED) {
        __bf16* sBF = (__bf16*)d_ws;
        __bf16* WqB = sBF + STATES_N;
        __bf16* WkB = WqB + W_N;
        __bf16* WvB = WkB + W_N;
        hipLaunchKernelGGL(cvt_f32_bf16, dim3(STATES_N / 1024), dim3(256), 0, stream, states, sBF, STATES_N);
        hipLaunchKernelGGL(cvt_f32_bf16, dim3(W_N / 1024),      dim3(256), 0, stream, Wq, WqB, W_N);
        hipLaunchKernelGGL(cvt_f32_bf16, dim3(W_N / 1024),      dim3(256), 0, stream, Wk, WkB, W_N);
        hipLaunchKernelGGL(cvt_f32_bf16, dim3(W_N / 1024),      dim3(256), 0, stream, Wv, WvB, W_N);
        hipLaunchKernelGGL(fused_local_attn_async, grid, block, 0, stream,
                           sBF, WqB, bq, WkB, bk, WvB, bv, out);
    } else {
        hipLaunchKernelGGL(fused_local_attn_fb, grid, block, 0, stream,
                           states, Wq, bq, Wk, bk, Wv, bv, out);
    }
}
